// DGCNGRU_20572893347929
// MI455X (gfx1250) — matrix operand order repaired and checked by reading, not compiled, not run on hardware
//
#include <hip/hip_runtime.h>
#include <hip/hip_bf16.h>

#define N_MSG 131072
#define MAX_NEI 6
#define IN_SZ 128
#define H_SZ 128
#define DEPTH 4

#define MT 32              // messages per block
#define HNS (H_SZ + 8)     // padded LDS stride (bf16) for gathered neighbor rows
#define SS  (H_SZ + 4)     // padded LDS stride (f32) for sum_h / sum_gated

typedef __attribute__((ext_vector_type(16))) __bf16 v16bf;
typedef __attribute__((ext_vector_type(8)))  __bf16 v8bf;
typedef __attribute__((ext_vector_type(8)))  float  v8f;

static __device__ __forceinline__ float sigmoidf_(float x) {
  return 1.0f / (1.0f + __expf(-x));
}

// A-fragment (16-bit, 16x32): lane<16 holds K 0..15, lane>=16 holds K 16..31 (contiguous)
static __device__ __forceinline__ v16bf make_a(const __bf16* lo, const __bf16* hi) {
  v8bf l = *(const v8bf*)lo;
  v8bf h = *(const v8bf*)hi;
  v16bf r;
#pragma unroll
  for (int i = 0; i < 8; ++i) { r[i] = l[i]; r[i + 8] = h[i]; }
  return r;
}

// A-fragment built from fp32 LDS tiles (convert to bf16 on load)
static __device__ __forceinline__ v16bf cvt_a(const float* lo, const float* hi) {
  v16bf r;
#pragma unroll
  for (int i = 0; i < 8; ++i) { r[i] = (__bf16)lo[i]; r[i + 8] = (__bf16)hi[i]; }
  return r;
}

static __device__ __forceinline__ v8f wmma_bf16(v16bf a, v16bf b, v8f c) {
  return __builtin_amdgcn_wmma_f32_16x16x32_bf16(false, a, false, b, (short)0, c,
                                                 false, false);
}

// ---- prep: zero h0 (bf16) and convert fmess -> bf16 -------------------------
__global__ void prep_main_kernel(const float* __restrict__ fmess,
                                 __bf16* __restrict__ fmess_bf,
                                 __bf16* __restrict__ h0) {
  for (int i = blockIdx.x * blockDim.x + threadIdx.x; i < N_MSG * H_SZ;
       i += gridDim.x * blockDim.x) {
    fmess_bf[i] = (__bf16)fmess[i];
    h0[i] = (__bf16)0.0f;
  }
}

// ---- prep: convert weights -> bf16 -----------------------------------------
__global__ void prep_w_kernel(const float* __restrict__ Wz, const float* __restrict__ Wh,
                              const float* __restrict__ Ur, const float* __restrict__ Wr,
                              __bf16* __restrict__ Wzb, __bf16* __restrict__ Whb,
                              __bf16* __restrict__ Urb, __bf16* __restrict__ Wrb) {
  int i = blockIdx.x * blockDim.x + threadIdx.x;  // 0..32767
  Wzb[i] = (__bf16)Wz[i];
  Whb[i] = (__bf16)Wh[i];
  if (i < H_SZ * H_SZ) {
    Urb[i] = (__bf16)Ur[i];
    Wrb[i] = (__bf16)Wr[i];
  }
}

// ---- r_x = fmess @ Wr^T  (WMMA, M=131072 N=128 K=128) ----------------------
__global__ __launch_bounds__(128) void rx_kernel(const __bf16* __restrict__ fm,
                                                 const __bf16* __restrict__ Wr,
                                                 float* __restrict__ rx) {
  const int wave = threadIdx.x >> 5;
  const int lane = threadIdx.x & 31;
  const int lhalf = lane >> 4;
  const int lmod = lane & 15;
  const int m0 = blockIdx.x * 64;
#pragma unroll
  for (int cti = 0; cti < 2; ++cti) {
    const int ct = wave * 2 + cti;
    const int n = ct * 16 + lmod;
    v16bf b[4];
#pragma unroll
    for (int kt = 0; kt < 4; ++kt)   // B[k][n] = Wr[n][k]: 16 contiguous K per lane
      b[kt] = *(const v16bf*)(Wr + n * IN_SZ + kt * 32 + lhalf * 16);
#pragma unroll
    for (int mt = 0; mt < 4; ++mt) {
      const __bf16* ap = fm + (m0 + mt * 16 + lmod) * IN_SZ;
      v8f acc = {};
#pragma unroll
      for (int kt = 0; kt < 4; ++kt) {
        v16bf a = make_a(ap + kt * 32 + lhalf * 16, ap + kt * 32 + lhalf * 16 + 8);
        acc = wmma_bf16(a, b[kt], acc);
      }
#pragma unroll
      for (int r = 0; r < 8; ++r) {
        int m = m0 + mt * 16 + r + 8 * lhalf;  // C layout: M = r + 8*(lane/16)
        rx[m * H_SZ + n] = acc[r];
      }
    }
  }
}

// ---- one GRU step, fully fused ---------------------------------------------
__global__ __launch_bounds__(128) void step_kernel(
    const __bf16* __restrict__ h_cur, __bf16* __restrict__ h_next,
    const __bf16* __restrict__ fm, const int* __restrict__ bgraph,
    const float* __restrict__ rx,
    const __bf16* __restrict__ Wz, const __bf16* __restrict__ Wh,
    const __bf16* __restrict__ Ur,
    const float* __restrict__ Wz_b, const float* __restrict__ Ur_b,
    const float* __restrict__ Wh_b,
    float* __restrict__ out_f32) {
  __shared__ __bf16 hn[MAX_NEI * MT * HNS];   // gathered neighbor h (nei-major)
  __shared__ float sumh[MT * SS];
  __shared__ float sumg[MT * SS];

  const int tid = threadIdx.x;
  const int m0 = blockIdx.x * MT;

  // phase 1: gather 192 rows (256B each) of h_cur into LDS, coalesced uint4 bursts
  {
    const uint4* hsrc = (const uint4*)h_cur;  // 16 x uint4 per 128-bf16 row
    for (int e = tid; e < MAX_NEI * MT * 16; e += 128) {
      int row = e >> 4;
      int u = e & 15;
      int nei = row / MT;
      int m = row - nei * MT;
      int idx = bgraph[(m0 + m) * MAX_NEI + nei];
      *(uint4*)(&hn[row * HNS + u * 8]) = hsrc[idx * 16 + u];
    }
  }
  __syncthreads();

  // phase 2: sum over neighbors (fp32), zero gated accumulator
  for (int j = tid; j < MT * H_SZ; j += 128) {
    int m = j >> 7;
    int k = j & 127;
    float s = 0.0f;
#pragma unroll
    for (int nei = 0; nei < MAX_NEI; ++nei)
      s += (float)hn[(nei * MT + m) * HNS + k];
    sumh[m * SS + k] = s;
    sumg[m * SS + k] = 0.0f;
  }
  __syncthreads();

  const int wave = tid >> 5, lane = tid & 31;
  const int lhalf = lane >> 4, lmod = lane & 15;

  // phase 3: r = sigmoid(rx + h_nei @ Ur^T + b); sum_gated += r * h_nei
  // nei-major row tiling => every (m,n) LDS RMW is owned by exactly one lane.
#pragma unroll
  for (int cti = 0; cti < 2; ++cti) {
    const int ct = wave * 2 + cti;
    const int n = ct * 16 + lmod;
    v16bf b[4];
#pragma unroll
    for (int kt = 0; kt < 4; ++kt)
      b[kt] = *(const v16bf*)(Ur + n * H_SZ + kt * 32 + lhalf * 16);
    const float urb = Ur_b[n];
    for (int rt = 0; rt < MAX_NEI * (MT / 16); ++rt) {  // 12 row tiles
      const int nei = rt >> 1;
      const int mt = rt & 1;
      const __bf16* ap = &hn[(nei * MT + mt * 16 + lmod) * HNS];
      v8f acc = {};
#pragma unroll
      for (int kt = 0; kt < 4; ++kt) {
        v16bf a = make_a(ap + kt * 32 + lhalf * 16, ap + kt * 32 + lhalf * 16 + 8);
        acc = wmma_bf16(a, b[kt], acc);
      }
#pragma unroll
      for (int r = 0; r < 8; ++r) {
        int m = mt * 16 + r + 8 * lhalf;
        float rv = sigmoidf_(rx[(m0 + m) * H_SZ + n] + acc[r] + urb);
        float hv = (float)hn[(nei * MT + m) * HNS + n];
        sumg[m * SS + n] += rv * hv;
      }
    }
  }
  __syncthreads();

  // phase 4: z / pre_h GEMMs (shared fmess A-fragments) + fused blend
#pragma unroll
  for (int cti = 0; cti < 2; ++cti) {
    const int ct = wave * 2 + cti;
    const int n = ct * 16 + lmod;
    const float zb = Wz_b[n];
    const float hb = Wh_b[n];
#pragma unroll
    for (int mt = 0; mt < MT / 16; ++mt) {
      v8f az = {}, ah = {};
      const __bf16* fap = fm + (m0 + mt * 16 + lmod) * IN_SZ;
#pragma unroll
      for (int kt = 0; kt < 4; ++kt) {  // K 0..127 : fmess part of concat
        v16bf a = make_a(fap + kt * 32 + lhalf * 16, fap + kt * 32 + lhalf * 16 + 8);
        v16bf bz = *(const v16bf*)(Wz + n * 256 + kt * 32 + lhalf * 16);
        v16bf bh = *(const v16bf*)(Wh + n * 256 + kt * 32 + lhalf * 16);
        az = wmma_bf16(a, bz, az);
        ah = wmma_bf16(a, bh, ah);
      }
      const float* shp = &sumh[(mt * 16 + lmod) * SS];
      const float* sgp = &sumg[(mt * 16 + lmod) * SS];
#pragma unroll
      for (int kt = 0; kt < 4; ++kt) {  // K 128..255 : sum_h / sum_gated part
        v16bf aZ = cvt_a(shp + kt * 32 + lhalf * 16, shp + kt * 32 + lhalf * 16 + 8);
        v16bf aH = cvt_a(sgp + kt * 32 + lhalf * 16, sgp + kt * 32 + lhalf * 16 + 8);
        v16bf bz = *(const v16bf*)(Wz + n * 256 + 128 + kt * 32 + lhalf * 16);
        v16bf bh = *(const v16bf*)(Wh + n * 256 + 128 + kt * 32 + lhalf * 16);
        az = wmma_bf16(aZ, bz, az);
        ah = wmma_bf16(aH, bh, ah);
      }
#pragma unroll
      for (int r = 0; r < 8; ++r) {
        int ml = mt * 16 + r + 8 * lhalf;
        int gm = m0 + ml;
        float z = sigmoidf_(az[r] + zb);
        float ph = tanhf(ah[r] + hb);
        float sh = sumh[ml * SS + n];
        float nv = (1.0f - z) * sh + z * ph;
        if (gm == 0) nv = 0.0f;  // padding mask
        h_next[gm * H_SZ + n] = (__bf16)nv;
        if (out_f32) out_f32[gm * H_SZ + n] = nv;
      }
    }
  }
}

extern "C" void kernel_launch(void* const* d_in, const int* in_sizes, int n_in,
                              void* d_out, int out_size, void* d_ws, size_t ws_size,
                              hipStream_t stream) {
  const float* fmess  = (const float*)d_in[0];
  const int*   bgraph = (const int*)d_in[1];
  // d_in[2] = depth (== 4, compile-time constant DEPTH)
  const float* Wz   = (const float*)d_in[3];
  const float* Wz_b = (const float*)d_in[4];
  const float* Wr   = (const float*)d_in[5];
  const float* Ur   = (const float*)d_in[6];
  const float* Ur_b = (const float*)d_in[7];
  const float* Wh   = (const float*)d_in[8];
  const float* Wh_b = (const float*)d_in[9];
  float* out = (float*)d_out;

  char* ws = (char*)d_ws;
  size_t off = 0;
  __bf16* h_a   = (__bf16*)(ws + off); off += (size_t)N_MSG * H_SZ * 2;
  __bf16* h_b   = (__bf16*)(ws + off); off += (size_t)N_MSG * H_SZ * 2;
  float*  rx    = (float*) (ws + off); off += (size_t)N_MSG * H_SZ * 4;
  __bf16* fm_bf = (__bf16*)(ws + off); off += (size_t)N_MSG * IN_SZ * 2;
  __bf16* Wz_bf = (__bf16*)(ws + off); off += (size_t)H_SZ * 256 * 2;
  __bf16* Wh_bf = (__bf16*)(ws + off); off += (size_t)H_SZ * 256 * 2;
  __bf16* Ur_bf = (__bf16*)(ws + off); off += (size_t)H_SZ * H_SZ * 2;
  __bf16* Wr_bf = (__bf16*)(ws + off); off += (size_t)H_SZ * H_SZ * 2;

  prep_main_kernel<<<4096, 256, 0, stream>>>(fmess, fm_bf, h_a);
  prep_w_kernel<<<(H_SZ * 256) / 256, 256, 0, stream>>>(Wz, Wh, Ur, Wr,
                                                        Wz_bf, Wh_bf, Ur_bf, Wr_bf);
  rx_kernel<<<N_MSG / 64, 128, 0, stream>>>(fm_bf, Wr_bf, rx);

  __bf16* hc = h_a;
  __bf16* hnx = h_b;
  for (int d = 0; d < DEPTH; ++d) {
    float* o = (d == DEPTH - 1) ? out : nullptr;
    step_kernel<<<N_MSG / MT, 128, 0, stream>>>(hc, hnx, fm_bf, bgraph, rx,
                                                Wz_bf, Wh_bf, Ur_bf,
                                                Wz_b, Ur_b, Wh_b, o);
    __bf16* t = hc; hc = hnx; hnx = t;
  }
}
